// VMLA_Block_38963943309472
// MI455X (gfx1250) — compile-verified, matrix operand order checked by reading, not run on hardware
//
#include <hip/hip_runtime.h>
#include <hip/hip_bf16.h>
#include <math.h>

// ---------------------------------------------------------------------------
// Types for CDNA5 WMMA (POD ext-vector types safe for unions / bit casts)
// ---------------------------------------------------------------------------
typedef __attribute__((ext_vector_type(16))) __bf16        bf16x16;
typedef __attribute__((ext_vector_type(8)))  float         floatx8;
typedef __attribute__((ext_vector_type(4)))  float         floatx4;
typedef __attribute__((ext_vector_type(4)))  unsigned int  uint32x4;
typedef __attribute__((ext_vector_type(2)))  unsigned int  uint32x2;

#define TP 40  // LDS pitch in bf16 elems: 80 bytes -> every 8-elem run 16B aligned

// Load one 16-bit A/B WMMA fragment (two 16B runs) from LDS via ds_load_b128.
__device__ __forceinline__ bf16x16 ld_frag(const __bf16* p)
{
    union { bf16x16 v; uint32x4 q[2]; } u;
    u.q[0] = *(const uint32x4*)(p);
    u.q[1] = *(const uint32x4*)(p + 16);
    return u.v;
}

// Pack 4 floats into 4 bf16 hi parts + 4 bf16 lo parts (3x-bf16 split).
__device__ __forceinline__ void split4(const floatx4 x, uint32x2& hi, uint32x2& lo)
{
    union { __bf16 h[4]; uint32x2 u; } hh, ll;
    #pragma unroll
    for (int j = 0; j < 4; ++j) {
        __bf16 hb = (__bf16)x[j];
        hh.h[j] = hb;
        ll.h[j] = (__bf16)(x[j] - (float)hb);
    }
    hi = hh.u;
    lo = ll.u;
}

// ---------------------------------------------------------------------------
// Generic batched GEMM:  C[b] = A[b] (MxK) @ op(B[b]) (KxN)  [+ epilogue]
//   BT = false : B is KxN row-major;  BT = true : B is NxK row-major (A@B^T)
//   EPI: 0 none, 1 exact GELU, 2 add D (residual, same ldc)
// 3x-bf16 split -> fp32-class accuracy, 3 WMMA per 16x16x32 tile.
// Block 128 (4 waves), workgroup tile 64x64, wave tile 32x32 (2x2 C tiles,
// 12 WMMA per K-step), K step 32.  Requires M%64==0, N%64==0, K%32==0.
// ---------------------------------------------------------------------------
template <bool BT, int EPI>
__global__ __launch_bounds__(128)
void gemm_bf16x3(const float* __restrict__ A, long long sA, int lda,
                 const float* __restrict__ Bm, long long sB, int ldb,
                 float* __restrict__ C, long long sC, int ldc,
                 const float* __restrict__ Dm, long long sD,
                 int M, int N, int K)
{
    __shared__ __align__(16) __bf16 sAh[64 * TP], sAl[64 * TP];
    __shared__ __align__(16) __bf16 sBh[64 * TP], sBl[64 * TP];
    (void)M; (void)N;

    const int b = blockIdx.z;
    A  += (long long)b * sA;
    Bm += (long long)b * sB;
    C  += (long long)b * sC;
    if (EPI == 2) Dm += (long long)b * sD;

    const int bm = blockIdx.y * 64;
    const int bn = blockIdx.x * 64;
    const int t = threadIdx.x, lane = t & 31, wave = t >> 5;
    const int wr = (wave >> 1) * 32;   // wave row offset in 64x64 tile
    const int wc = (wave & 1) * 32;    // wave col offset
    const int l15 = lane & 15;
    const int k0  = (lane & 16) ? 8 : 0;
    const int mb  = (lane & 16) ? 8 : 0;

    floatx8 acc[2][2];
    #pragma unroll
    for (int mi = 0; mi < 2; ++mi)
        #pragma unroll
        for (int ni = 0; ni < 2; ++ni)
            acc[mi][ni] = floatx8{0.f,0.f,0.f,0.f,0.f,0.f,0.f,0.f};

    for (int kt = 0; kt < K; kt += 32) {
        // prefetch next K tile (global_prefetch_b8)
        if (kt + 32 < K) {
            __builtin_prefetch(&A[(long long)(bm + (t >> 1)) * lda + kt + 32 + (t & 1) * 16], 0, 1);
            if (BT) __builtin_prefetch(&Bm[(long long)(bn + (t >> 1)) * ldb + kt + 32 + (t & 1) * 16], 0, 1);
            else    __builtin_prefetch(&Bm[(long long)(kt + 32 + (t >> 2)) * ldb + bn + (t & 3) * 16], 0, 1);
        }

        // ---- stage A tile 64x32: float4 global reads, b64 packed LDS stores ----
        #pragma unroll
        for (int i = 0; i < 4; ++i) {
            int qi = t + i * 128;               // 0..511 quads
            int r = qi >> 3, c4 = (qi & 7) * 4; // r: m row, c4: k col (x4)
            floatx4 av = *(const floatx4*)&A[(long long)(bm + r) * lda + (kt + c4)];
            uint32x2 hi, lo;
            split4(av, hi, lo);
            *(uint32x2*)&sAh[r * TP + c4] = hi;
            *(uint32x2*)&sAl[r * TP + c4] = lo;
        }
        // ---- stage B tile as [n][k] ----
        if (BT) {
            #pragma unroll
            for (int i = 0; i < 4; ++i) {
                int qi = t + i * 128;
                int r = qi >> 3, c4 = (qi & 7) * 4;   // r: n, c4: k
                floatx4 bv = *(const floatx4*)&Bm[(long long)(bn + r) * ldb + (kt + c4)];
                uint32x2 hi, lo;
                split4(bv, hi, lo);
                *(uint32x2*)&sBh[r * TP + c4] = hi;
                *(uint32x2*)&sBl[r * TP + c4] = lo;
            }
        } else {
            #pragma unroll
            for (int i = 0; i < 4; ++i) {
                int qi = t + i * 128;
                int k = qi >> 4, n4 = (qi & 15) * 4;  // contiguous read along n
                floatx4 bv = *(const floatx4*)&Bm[(long long)(kt + k) * ldb + (bn + n4)];
                #pragma unroll
                for (int j = 0; j < 4; ++j) {
                    __bf16 hb = (__bf16)bv[j];
                    sBh[(n4 + j) * TP + k] = hb;
                    sBl[(n4 + j) * TP + k] = (__bf16)(bv[j] - (float)hb);
                }
            }
        }
        __syncthreads();

        // ---- fragments (ds_load_b128 pairs) ----
        bf16x16 a_h[2], a_l[2], b_h[2], b_l[2];
        #pragma unroll
        for (int x = 0; x < 2; ++x) {
            a_h[x] = ld_frag(&sAh[(wr + x * 16 + l15) * TP + k0]);
            a_l[x] = ld_frag(&sAl[(wr + x * 16 + l15) * TP + k0]);
            b_h[x] = ld_frag(&sBh[(wc + x * 16 + l15) * TP + k0]);
            b_l[x] = ld_frag(&sBl[(wc + x * 16 + l15) * TP + k0]);
        }
        // ---- 2x2 tiles x 3 WMMA ----
        #pragma unroll
        for (int mi = 0; mi < 2; ++mi) {
            #pragma unroll
            for (int ni = 0; ni < 2; ++ni) {
                acc[mi][ni] = __builtin_amdgcn_wmma_f32_16x16x32_bf16(false, a_h[mi], false, b_h[ni], (short)0, acc[mi][ni], false, false);
                acc[mi][ni] = __builtin_amdgcn_wmma_f32_16x16x32_bf16(false, a_h[mi], false, b_l[ni], (short)0, acc[mi][ni], false, false);
                acc[mi][ni] = __builtin_amdgcn_wmma_f32_16x16x32_bf16(false, a_l[mi], false, b_h[ni], (short)0, acc[mi][ni], false, false);
            }
        }
        __syncthreads();
    }

    // ---- epilogue + store (C tile: lane -> col, vgpr -> row) ----
    #pragma unroll
    for (int mi = 0; mi < 2; ++mi) {
        #pragma unroll
        for (int ni = 0; ni < 2; ++ni) {
            const int m0  = bm + wr + mi * 16 + mb;
            const int n_g = bn + wc + ni * 16 + l15;
            #pragma unroll
            for (int v = 0; v < 8; ++v) {
                float x = acc[mi][ni][v];
                if (EPI == 1) x = 0.5f * x * (1.0f + erff(x * 0.70710678118654752f));
                if (EPI == 2) x += Dm[(long long)(m0 + v) * ldc + n_g];
                C[(long long)(m0 + v) * ldc + n_g] = x;
            }
        }
    }
}

// ---------------------------------------------------------------------------
// LayerNorm over last dim D (one block of 256 per row)
// ---------------------------------------------------------------------------
__global__ __launch_bounds__(256)
void ln_kernel(const float* __restrict__ X, const float* __restrict__ G,
               float* __restrict__ Y, int D)
{
    __shared__ float red[8];
    const long long row = blockIdx.x;
    const float* x = X + row * D;
    float* y = Y + row * D;
    const int t = threadIdx.x;

    float s = 0.f;
    for (int i = t; i < D; i += 256) s += x[i];
    #pragma unroll
    for (int o = 16; o > 0; o >>= 1) s += __shfl_xor(s, o, 32);
    if ((t & 31) == 0) red[t >> 5] = s;
    __syncthreads();
    if (t == 0) { float tt = 0.f; for (int i = 0; i < 8; ++i) tt += red[i]; red[0] = tt; }
    __syncthreads();
    const float mean = red[0] / (float)D;
    __syncthreads();

    float v = 0.f;
    for (int i = t; i < D; i += 256) { float d = x[i] - mean; v += d * d; }
    #pragma unroll
    for (int o = 16; o > 0; o >>= 1) v += __shfl_xor(v, o, 32);
    if ((t & 31) == 0) red[t >> 5] = v;
    __syncthreads();
    if (t == 0) { float tt = 0.f; for (int i = 0; i < 8; ++i) tt += red[i]; red[0] = tt; }
    __syncthreads();
    const float rstd = rsqrtf(red[0] / (float)D + 1e-6f);

    for (int i = t; i < D; i += 256) y[i] = (x[i] - mean) * rstd * G[i];
}

// ---------------------------------------------------------------------------
// Build q/k head tensor (B,H,SN,64): first 32 = content part, last 32 = RoPE'd
// rotary part.  Z,R: (B,SN,H*32) ; F: (SN,16)
// ---------------------------------------------------------------------------
__global__ __launch_bounds__(256)
void rope_concat(const float* __restrict__ Z, const float* __restrict__ R,
                 const float* __restrict__ F, float* __restrict__ O)
{
    const int idx = blockIdx.x * 256 + threadIdx.x;   // B*H*SN*64 threads
    const int d = idx & 63;
    const int n = (idx >> 6) & 1023;
    const int h = (idx >> 16) & 15;
    const int b = idx >> 20;
    float out;
    if (d < 32) {
        out = Z[(((long long)b * 1024 + n) * 512) + h * 32 + d];
    } else {
        const int dr = d - 32;
        const float* rrow = R + (((long long)b * 1024 + n) * 512) + h * 32;
        const float xv  = rrow[dr];
        const float rot = (dr < 16) ? -rrow[dr + 16] : rrow[dr - 16];
        const float f = F[n * 16 + (dr & 15)];
        out = xv * cosf(f) + rot * sinf(f);
    }
    O[idx] = out;
}

// ---------------------------------------------------------------------------
// Flash attention with additive mask.
//   Q,K: (B*H, SN, 64) ; V,O: (B, SN, H*64) [head slice h*64] ; MM: (B,SN,SN)
// Grid (SN/64, B*H), block 128 (4 waves).  Each wave: 16 q rows.
// Key chunks of 32, online softmax, WMMA bf16x3 for both QK^T and PV.
// ---------------------------------------------------------------------------
__global__ __launch_bounds__(128)
void attention_kernel(const float* __restrict__ Q, const float* __restrict__ Km,
                      const float* __restrict__ V, const float* __restrict__ MM,
                      float* __restrict__ O)
{
    constexpr int VP = 40;
    __shared__ __align__(16) __bf16 sVh[64 * VP], sVl[64 * VP];
    __shared__ float sP[4][16][36];

    const int bh = blockIdx.y, b = bh >> 4, h = bh & 15;
    const int t = threadIdx.x, lane = t & 31, wave = t >> 5;
    const int l15 = lane & 15;
    const int mb = (lane & 16) ? 8 : 0;
    const int k0 = (lane & 16) ? 8 : 0;
    const int r0 = blockIdx.x * 64 + wave * 16;

    // q fragments (rows r0+l15, K-dim = 64 head dims as two 32-chunks)
    bf16x16 qh[2], ql[2];
    {
        const float* qrow = Q + ((long long)bh * 1024 + (r0 + l15)) * 64;
        #pragma unroll
        for (int c = 0; c < 2; ++c) {
            #pragma unroll
            for (int i = 0; i < 8; ++i) {
                float x0 = qrow[c * 32 + k0 + i];
                float x1 = qrow[c * 32 + k0 + 16 + i];
                __bf16 h0 = (__bf16)x0, h1 = (__bf16)x1;
                qh[c][i] = h0;  qh[c][8 + i] = h1;
                ql[c][i]     = (__bf16)(x0 - (float)h0);
                ql[c][8 + i] = (__bf16)(x1 - (float)h1);
            }
        }
    }

    float m_run[8], l_run[8];
    floatx8 accO[4];
    #pragma unroll
    for (int v = 0; v < 8; ++v) { m_run[v] = -1e30f; l_run[v] = 0.f; }
    #pragma unroll
    for (int j = 0; j < 4; ++j) accO[j] = floatx8{0.f,0.f,0.f,0.f,0.f,0.f,0.f,0.f};

    const float scale = 0.125f;  // 1/sqrt(64)

    for (int kt = 0; kt < 32; ++kt) {          // 32 key-chunks of 32
        __syncthreads();
        // stage V chunk transposed: sV[d][k], d=0..63, k=0..31
        #pragma unroll
        for (int i = 0; i < 16; ++i) {
            int e = t + i * 128;               // 0..2047
            int kk = e >> 6, d = e & 63;       // coalesced over d
            float x = V[((long long)b * 1024 + (kt * 32 + kk)) * 1024 + h * 64 + d];
            __bf16 xh = (__bf16)x;
            sVh[d * VP + kk] = xh;
            sVl[d * VP + kk] = (__bf16)(x - (float)xh);
        }
        __syncthreads();

        // --- scores: two 16x16 tiles (j=0,1) ---
        floatx8 sc[2];
        #pragma unroll
        for (int j = 0; j < 2; ++j) {
            const int key = kt * 32 + j * 16 + l15;
            const float* krow = Km + ((long long)bh * 1024 + key) * 64;
            floatx8 a = {0.f,0.f,0.f,0.f,0.f,0.f,0.f,0.f};
            #pragma unroll
            for (int c = 0; c < 2; ++c) {
                bf16x16 bfh, bfl;
                #pragma unroll
                for (int i = 0; i < 8; ++i) {
                    float x0 = krow[c * 32 + k0 + i];
                    float x1 = krow[c * 32 + k0 + 16 + i];
                    __bf16 h0 = (__bf16)x0, h1 = (__bf16)x1;
                    bfh[i] = h0;  bfh[8 + i] = h1;
                    bfl[i]     = (__bf16)(x0 - (float)h0);
                    bfl[8 + i] = (__bf16)(x1 - (float)h1);
                }
                a = __builtin_amdgcn_wmma_f32_16x16x32_bf16(false, qh[c], false, bfh, (short)0, a, false, false);
                a = __builtin_amdgcn_wmma_f32_16x16x32_bf16(false, qh[c], false, bfl, (short)0, a, false, false);
                a = __builtin_amdgcn_wmma_f32_16x16x32_bf16(false, ql[c], false, bfh, (short)0, a, false, false);
            }
            #pragma unroll
            for (int v = 0; v < 8; ++v)
                a[v] = a[v] * scale +
                       MM[((long long)b * 1024 + (r0 + mb + v)) * 1024 + (kt * 32 + j * 16 + l15)];
            sc[j] = a;
        }

        // --- online softmax update (row stats replicated across 16-lane half) ---
        #pragma unroll
        for (int v = 0; v < 8; ++v) {
            float mx = fmaxf(sc[0][v], sc[1][v]);
            #pragma unroll
            for (int o = 8; o > 0; o >>= 1) mx = fmaxf(mx, __shfl_xor(mx, o, 32));
            float m_new = fmaxf(m_run[v], mx);
            float fac = __expf(m_run[v] - m_new);
            float p0 = __expf(sc[0][v] - m_new);
            float p1 = __expf(sc[1][v] - m_new);
            float rs = p0 + p1;
            #pragma unroll
            for (int o = 8; o > 0; o >>= 1) rs += __shfl_xor(rs, o, 32);
            l_run[v] = l_run[v] * fac + rs;
            m_run[v] = m_new;
            #pragma unroll
            for (int j = 0; j < 4; ++j) accO[j][v] *= fac;
            sP[wave][mb + v][l15]      = p0;   // redistribute C-layout -> A-layout
            sP[wave][mb + v][16 + l15] = p1;
        }
        asm volatile("s_wait_dscnt 0x0" ::: "memory");

        // --- P fragments (A layout: row = l15, k runs over 32 keys) ---
        bf16x16 ph, pl;
        #pragma unroll
        for (int i = 0; i < 8; ++i) {
            float x0 = sP[wave][l15][k0 + i];
            float x1 = sP[wave][l15][k0 + 16 + i];
            __bf16 h0 = (__bf16)x0, h1 = (__bf16)x1;
            ph[i] = h0;  ph[8 + i] = h1;
            pl[i]     = (__bf16)(x0 - (float)h0);
            pl[8 + i] = (__bf16)(x1 - (float)h1);
        }
        // --- accO += P @ Vchunk (4 d-tiles of 16) ---
        #pragma unroll
        for (int j = 0; j < 4; ++j) {
            const int d = j * 16 + l15;
            bf16x16 vh = ld_frag(&sVh[d * VP + k0]);
            bf16x16 vl = ld_frag(&sVl[d * VP + k0]);
            accO[j] = __builtin_amdgcn_wmma_f32_16x16x32_bf16(false, ph, false, vh, (short)0, accO[j], false, false);
            accO[j] = __builtin_amdgcn_wmma_f32_16x16x32_bf16(false, ph, false, vl, (short)0, accO[j], false, false);
            accO[j] = __builtin_amdgcn_wmma_f32_16x16x32_bf16(false, pl, false, vh, (short)0, accO[j], false, false);
        }
    }

    // --- finalize: divide by l, write (B,SN,H*64) ---
    #pragma unroll
    for (int j = 0; j < 4; ++j) {
        const int d = j * 16 + l15;
        #pragma unroll
        for (int v = 0; v < 8; ++v) {
            const int row = r0 + mb + v;
            O[((long long)b * 1024 + row) * 1024 + h * 64 + d] = accO[j][v] / l_run[v];
        }
    }
}

// ---------------------------------------------------------------------------
// Host orchestration.  Workspace layout (floats, MF = 1<<20), lifetimes
// overlapped; peak = 44*MF floats = 176 MB.
// ---------------------------------------------------------------------------
extern "C" void kernel_launch(void* const* d_in, const int* in_sizes, int n_in,
                              void* d_out, int out_size, void* d_ws, size_t ws_size,
                              hipStream_t stream)
{
    (void)in_sizes; (void)n_in; (void)out_size; (void)ws_size;

    const float* input_q = (const float*)d_in[0];
    const float* ln_q_w  = (const float*)d_in[1];
    const float* ln_kv_w = (const float*)d_in[2];
    const float* te_q_w  = (const float*)d_in[3];
    const float* te_kv_w = (const float*)d_in[4];
    const float* enc_q_w = (const float*)d_in[5];
    const float* enc_kv_w= (const float*)d_in[6];
    const float* t_qz_w  = (const float*)d_in[7];
    const float* t_kz_w  = (const float*)d_in[8];
    const float* t_vz_w  = (const float*)d_in[9];
    const float* t_qr_w  = (const float*)d_in[10];
    const float* t_kr_w  = (const float*)d_in[11];
    const float* q_w     = (const float*)d_in[12];
    const float* k_w     = (const float*)d_in[13];
    const float* v_w     = (const float*)d_in[14];
    const float* qr_w    = (const float*)d_in[15];
    const float* kr_w    = (const float*)d_in[16];
    const float* mask_w1 = (const float*)d_in[17];
    const float* mask_w2 = (const float*)d_in[18];
    const float* rope_q_f= (const float*)d_in[19];
    const float* rope_k_f= (const float*)d_in[20];
    const float* out_w   = (const float*)d_in[21];
    const float* ln2_w   = (const float*)d_in[22];
    const float* mlp_w1  = (const float*)d_in[23];
    const float* mlp_w2  = (const float*)d_in[24];
    float* OUT = (float*)d_out;

    float* W = (float*)d_ws;
    const long long MF = 1ll << 20;
    float* xq    = W;             // 4MF; later: s (scores), later: mlph region
    float* s     = W;
    float* mlph  = W;             // 16MF spanning [0,16MF) when dead
    float* xkv   = W + 4 * MF;    // 4MF; later: q heads
    float* qbuf  = xkv;
    float* krs   = W + 8 * MF;    // 4MF; later: k heads
    float* kbuf  = krs;
    float* xqc   = W + 12 * MF;   // 1MF
    float* xattn = W + 12 * MF;   // 4MF (after mv* dead)
    float* xkvc  = W + 13 * MF;   // 1MF
    float* mvq   = W + 14 * MF;   // 1MF
    float* mvkv  = W + 15 * MF;   // 1MF
    float* qzp   = W + 16 * MF;   // 2MF; later: qrh
    float* qrh   = qzp;
    float* kzp   = W + 18 * MF;   // 2MF; later: krh
    float* krh   = kzp;
    float* vzp   = W + 20 * MF;   // 2MF
    float* qrs   = W + 22 * MF;   // 2MF
    float* qz    = W + 24 * MF;   // 2MF; later (with kz): x residual
    float* xres  = qz;
    float* kz    = W + 26 * MF;   // 2MF
    float* vz    = W + 28 * MF;   // 4MF; later: yln
    float* yln   = vz;
    float* maskh = W + 32 * MF;   // 8MF
    float* maskm = W + 40 * MF;   // 4MF

    const dim3 TB(128);

    // 1-2: LayerNorms of input
    ln_kernel<<<dim3(4096), dim3(256), 0, stream>>>(input_q, ln_q_w,  xq,  1024);
    ln_kernel<<<dim3(4096), dim3(256), 0, stream>>>(input_q, ln_kv_w, xkv, 1024);

    // 3-4: token mixing  xc[b] = te_w (SRxS) @ x[b] (SxD1)
    gemm_bf16x3<false,0><<<dim3(16, 4, 4), TB, 0, stream>>>(te_q_w,  0, 1024, xq,  MF, 1024, xqc,  262144, 1024, nullptr, 0, 256, 1024, 1024);
    gemm_bf16x3<false,0><<<dim3(16, 4, 4), TB, 0, stream>>>(te_kv_w, 0, 1024, xkv, MF, 1024, xkvc, 262144, 1024, nullptr, 0, 256, 1024, 1024);

    // 5-6: encoders  mv = xc @ enc_w^T
    gemm_bf16x3<true,0><<<dim3(16, 4, 4), TB, 0, stream>>>(xqc,  262144, 1024, enc_q_w,  0, 1024, mvq,  262144, 1024, nullptr, 0, 256, 1024, 1024);
    gemm_bf16x3<true,0><<<dim3(16, 4, 4), TB, 0, stream>>>(xkvc, 262144, 1024, enc_kv_w, 0, 1024, mvkv, 262144, 1024, nullptr, 0, 256, 1024, 1024);

    // 7-10: seq-lin expansions from z slices (ldb=1024 reads the [:512] slice)
    gemm_bf16x3<false,0><<<dim3(8, 16, 4), TB, 0, stream>>>(t_qz_w, 0, 256, mvq,  262144, 1024, qzp, 524288, 512, nullptr, 0, 1024, 512, 256);
    gemm_bf16x3<false,0><<<dim3(8, 16, 4), TB, 0, stream>>>(t_kz_w, 0, 256, mvkv, 262144, 1024, kzp, 524288, 512, nullptr, 0, 1024, 512, 256);
    gemm_bf16x3<false,0><<<dim3(8, 16, 4), TB, 0, stream>>>(t_vz_w, 0, 256, mvkv, 262144, 1024, vzp, 524288, 512, nullptr, 0, 1024, 512, 256);
    gemm_bf16x3<false,0><<<dim3(8, 16, 4), TB, 0, stream>>>(t_qr_w, 0, 256, mvq,  262144, 1024, qrs, 524288, 512, nullptr, 0, 1024, 512, 256);
    // 11: kr seq-lin from xkv
    gemm_bf16x3<false,0><<<dim3(16, 16, 4), TB, 0, stream>>>(t_kr_w, 0, 1024, xkv, MF, 1024, krs, MF, 1024, nullptr, 0, 1024, 1024, 1024);

    // 12-16: projections
    gemm_bf16x3<true,0><<<dim3(8, 16, 4), TB, 0, stream>>>(qzp, 524288, 512, q_w,  0, 512,  qz,  524288, 512,  nullptr, 0, 1024, 512, 512);
    gemm_bf16x3<true,0><<<dim3(8, 16, 4), TB, 0, stream>>>(kzp, 524288, 512, k_w,  0, 512,  kz,  524288, 512,  nullptr, 0, 1024, 512, 512);
    gemm_bf16x3<true,0><<<dim3(16, 16, 4), TB, 0, stream>>>(vzp, 524288, 512, v_w,  0, 512,  vz,  MF,     1024, nullptr, 0, 1024, 1024, 512);
    gemm_bf16x3<true,0><<<dim3(8, 16, 4), TB, 0, stream>>>(qrs, 524288, 512, qr_w, 0, 512,  qrh, 524288, 512,  nullptr, 0, 1024, 512, 512);
    gemm_bf16x3<true,0><<<dim3(8, 16, 4), TB, 0, stream>>>(krs, MF,    1024, kr_w, 0, 1024, krh, 524288, 512,  nullptr, 0, 1024, 512, 1024);

    // 17: content scores  s = qz @ kz^T  (batched B)
    gemm_bf16x3<true,0><<<dim3(16, 16, 4), TB, 0, stream>>>(qz, 524288, 512, kz, 524288, 512, s, MF, 1024, nullptr, 0, 1024, 1024, 512);

    // 18-19: mask MLP  mask = gelu(s @ w1^T) @ w2^T
    gemm_bf16x3<true,1><<<dim3(32, 16, 4), TB, 0, stream>>>(s,     MF,     1024, mask_w1, 0, 1024, maskh, 2 * MF, 2048, nullptr, 0, 1024, 2048, 1024);
    gemm_bf16x3<true,0><<<dim3(16, 16, 4), TB, 0, stream>>>(maskh, 2 * MF, 2048, mask_w2, 0, 2048, maskm, MF,     1024, nullptr, 0, 1024, 1024, 2048);

    // 20: build q/k head tensors with RoPE on rotary half
    rope_concat<<<dim3(16384), dim3(256), 0, stream>>>(qz, qrh, rope_q_f, qbuf);
    rope_concat<<<dim3(16384), dim3(256), 0, stream>>>(kz, krh, rope_k_f, kbuf);

    // 21: flash attention with additive mask
    attention_kernel<<<dim3(16, 64), TB, 0, stream>>>(qbuf, kbuf, vz, maskm, xattn);

    // 22: out proj + residual
    gemm_bf16x3<true,2><<<dim3(16, 16, 4), TB, 0, stream>>>(xattn, MF, 1024, out_w, 0, 1024, xres, MF, 1024, input_q, MF, 1024, 1024, 1024);

    // 23: LN2
    ln_kernel<<<dim3(4096), dim3(256), 0, stream>>>(xres, ln2_w, yln, 1024);

    // 24-25: MLP  out = gelu(y @ w1^T) @ w2^T + x
    gemm_bf16x3<true,1><<<dim3(64, 16, 4), TB, 0, stream>>>(yln,  MF,     1024, mlp_w1, 0, 1024, mlph, 4 * MF, 4096, nullptr, 0, 1024, 4096, 1024);
    gemm_bf16x3<true,2><<<dim3(16, 16, 4), TB, 0, stream>>>(mlph, 4 * MF, 4096, mlp_w2, 0, 4096, OUT,  MF,     1024, xres,    MF, 1024, 1024, 4096);
}
